// MDAFP_43611097923816
// MI455X (gfx1250) — matrix operand order repaired and checked by reading, not compiled, not run on hardware
//
#include <hip/hip_runtime.h>
#include <hip/hip_bf16.h>

typedef __attribute__((ext_vector_type(16))) __bf16 v16bf;
typedef __attribute__((ext_vector_type(8)))  float  v8f;

#define DEVFN static __device__ __forceinline__

constexpr int Bn = 4, Cc = 128, Hh = 128, Ww = 128, NHn = 8, CHd = 16;
constexpr int HW = Hh * Ww;                       // 16384
constexpr int Ld = 128;                           // axial sequence length
constexpr int Dd = 128 * CHd;                     // 2048 head dim after reshape
constexpr long long CHW = (long long)Cc * HW;     // 2,097,152
constexpr long long NEL = (long long)Bn * CHW;    // 8,388,608
constexpr float EPSV = 1e-5f;

DEVFN float sigmoidf(float x) { return 1.0f / (1.0f + __expf(-x)); }
DEVFN float siluf(float x)    { return x * sigmoidf(x); }

// ---------------------------------------------------------------------------
// Batched bf16 WMMA GEMM:  C[z] (+)= A[z] * B[z], fp32 accumulate.
// A: M x K row-major.  B: K x N row-major (TRANSB=false) or N x K (TRANSB=true).
// Block = 256 threads = 8 waves; wave w owns rows [16w,16w+16); block owns 64 cols.
// The 32(K) x 64(N) B chunk is shared by all 8 waves -> staged once in LDS per
// K-step with coalesced global loads; fragments then come from ds_load_b128.
// All launches use M as an exact multiple of 128 so EXEC is all-ones at WMMA.
// ---------------------------------------------------------------------------
template <bool TRANSB, bool ACC>
__global__ __launch_bounds__(256) void k_gemm_bf16(
    const __bf16* __restrict__ A, const __bf16* __restrict__ Bm,
    float* __restrict__ C, int M, int N, int K,
    long long sA, long long sB, long long sC)
{
  __shared__ __align__(16) __bf16 Bs[64][40];   // n-major, +8 pad (20-bank stride)

  const int tid  = threadIdx.x;
  const int lane = tid & 31;
  const int wave = tid >> 5;
  A  += (long long)blockIdx.z * sA;
  Bm += (long long)blockIdx.z * sB;
  C  += (long long)blockIdx.z * sC;

  const int mBase = blockIdx.y * 128 + wave * 16;
  const int nBase = blockIdx.x * 64;

  // cooperative stage mapping: 256 threads cover 64 n-rows x 32 k (8 per thread)
  const int sn = tid & 63;          // 0..63  (n within tile)
  const int sk = (tid >> 6) * 8;    // 0,8,16,24 (k chunk)

  v8f acc[4];
#pragma unroll
  for (int t = 0; t < 4; ++t)
#pragma unroll
    for (int r = 0; r < 8; ++r) acc[t][r] = 0.0f;

  const int rA     = lane & 15;          // row within A tile / col within B tile
  const int hiA    = (lane >> 4) & 1;    // lane half
  const int kaBase = hiA * 8;            // A fragment K base (16-bit layout)
  const int kbBase = hiA * 8;            // B fragment K base

  for (int k0 = 0; k0 < K; k0 += 32) {
    __syncthreads();                     // previous iteration's reads done
    if (TRANSB) {
      // B is N x K row-major: 16B contiguous per thread along K
      const __bf16* gp = Bm + (long long)(nBase + sn) * K + k0 + sk;
      *(uint4*)(&Bs[sn][sk]) = *(const uint4*)gp;
    } else {
      // B is K x N row-major: 64 threads read 64 consecutive n per k-row
      const __bf16* gp = Bm + (long long)(k0 + sk) * N + (nBase + sn);
#pragma unroll
      for (int j = 0; j < 8; ++j) Bs[sn][sk + j] = gp[(long long)j * N];
    }
    __syncthreads();

    // ---- A fragment: lane holds row mBase+rA, K = kaBase + {0..7, 16..23} ----
    v16bf a;
    const __bf16* ap = A + (long long)(mBase + rA) * K + k0 + kaBase;
    __builtin_prefetch(ap + 64, 0, 1);
#pragma unroll
    for (int i = 0; i < 16; ++i) a[i] = ap[(i < 8) ? i : (i + 8)];

#pragma unroll
    for (int t = 0; t < 4; ++t) {
      // ---- B fragment from LDS: lane holds col t*16+rA, K = kbBase + {0..7, 16..23} ----
      const __bf16* bp = &Bs[t * 16 + rA][kbBase];
      v16bf b;
#pragma unroll
      for (int i = 0; i < 16; ++i) b[i] = bp[(i < 8) ? i : (i + 8)];
      acc[t] = __builtin_amdgcn_wmma_f32_16x16x32_bf16(
          false, a, false, b, (short)0, acc[t], false, false);
    }
  }

#pragma unroll
  for (int t = 0; t < 4; ++t) {
    const int col = nBase + t * 16 + rA;
#pragma unroll
    for (int r = 0; r < 8; ++r) {
      const int row = mBase + r + hiA * 8;
      const long long idx = (long long)row * N + col;
      if (ACC) C[idx] += acc[t][r];
      else     C[idx]  = acc[t][r];
    }
  }
}

// ---------------------------------------------------------------------------
// LayerNorm over channel dim (per spatial position), biased variance.
// ---------------------------------------------------------------------------
__global__ void k_ln4d(const float* __restrict__ x, const float* __restrict__ w,
                       const float* __restrict__ b, float* __restrict__ out)
{
  const long long p = (long long)blockIdx.x * blockDim.x + threadIdx.x; // b*HW + sp
  const int bi = (int)(p / HW);
  const int sp = (int)(p % HW);
  const float* xb = x + (long long)bi * CHW + sp;
  float s = 0.f, ss = 0.f;
  for (int c = 0; c < Cc; ++c) { float v = xb[(long long)c * HW]; s += v; ss += v * v; }
  const float mu   = s / Cc;
  const float var  = ss / Cc - mu * mu;
  const float rstd = rsqrtf(var + EPSV);
  float* ob = out + (long long)bi * CHW + sp;
  for (int c = 0; c < Cc; ++c) {
    float v = xb[(long long)c * HW];
    ob[(long long)c * HW] = (v - mu) * rstd * w[c] + b[c];
  }
}

// ---------------------------------------------------------------------------
// Sum of six depthwise convs: (1xk) and (kx1) for k in {7,11,21}, SAME padding.
// ---------------------------------------------------------------------------
__global__ void k_dirmix(const float* __restrict__ in, float* __restrict__ out,
    const float* wh0, const float* wh1, const float* wh2,
    const float* bh0, const float* bh1, const float* bh2,
    const float* wv0, const float* wv1, const float* wv2,
    const float* bv0, const float* bv1, const float* bv2)
{
  const long long i = (long long)blockIdx.x * blockDim.x + threadIdx.x;
  if (i >= NEL) return;
  const int w = (int)(i % Ww);
  const int h = (int)((i / Ww) % Hh);
  const int c = (int)((i / HW) % Cc);
  const float* rowp = in + (i - w);
  const float* colp = in + (i - (long long)h * Ww);
  const float* WH[3] = {wh0, wh1, wh2};
  const float* BH[3] = {bh0, bh1, bh2};
  const float* WV[3] = {wv0, wv1, wv2};
  const float* BV[3] = {bv0, bv1, bv2};
  const int KS[3] = {7, 11, 21};
  float acc = 0.f;
  for (int t = 0; t < 3; ++t) {
    const int k = KS[t], half = k >> 1;
    const float* wp = WH[t] + c * k;
    for (int j = 0; j < k; ++j) {
      const int ww = w + j - half;
      if (ww >= 0 && ww < Ww) acc += wp[j] * rowp[ww];
    }
    acc += BH[t][c];
    const float* vp = WV[t] + c * k;
    for (int j = 0; j < k; ++j) {
      const int hh = h + j - half;
      if (hh >= 0 && hh < Hh) acc += vp[j] * colp[(long long)hh * Ww];
    }
    acc += BV[t][c];
  }
  out[i] = acc;
}

// ---------------------------------------------------------------------------
// fp32 -> bf16 conversion (bulk) and weight tile conversion (with row stride).
// ---------------------------------------------------------------------------
__global__ void k_cvt_bf16(const float* __restrict__ in, __bf16* __restrict__ out,
                           long long n)
{
  const long long i = (long long)blockIdx.x * blockDim.x + threadIdx.x;
  if (i < n) out[i] = (__bf16)in[i];
}

__global__ void k_wcvt(const float* __restrict__ src, __bf16* __restrict__ dst,
                       int rows, int cols, int ld, int coloff)
{
  const int i = blockIdx.x * blockDim.x + threadIdx.x;
  if (i < rows * cols) {
    const int r = i / cols, cI = i % cols;
    dst[i] = (__bf16)src[(long long)r * ld + coloff + cI];
  }
}

// ---------------------------------------------------------------------------
// Axial reshape + L2 normalize over the reshaped head dim (D=2048).
// axisW=0: (b,head,h,w*16+ci) ; axisW=1: (b,head,w,h*16+ci)
// Writes normalized bf16 (for Q/K) and optional raw bf16 (for V).
// ---------------------------------------------------------------------------
__global__ void k_resh_l2n(const float* __restrict__ src, __bf16* __restrict__ qn,
                           __bf16* __restrict__ raw, int axisW)
{
  __shared__ float red[256];
  const int row  = blockIdx.x;                  // 0 .. Bn*NHn*Ld
  const int b    = row / (NHn * Ld);
  const int head = (row / Ld) % NHn;
  const int pos  = row % Ld;
  const int tid  = threadIdx.x;
  float vals[8];
  float ss = 0.f;
#pragma unroll
  for (int u = 0; u < 8; ++u) {
    const int d  = u * 256 + tid;
    const int ci = d & 15, t2 = d >> 4;
    const long long si = axisW
        ? ((long long)(b * Cc + head * CHd + ci) * Hh + t2) * Ww + pos
        : ((long long)(b * Cc + head * CHd + ci) * Hh + pos) * Ww + t2;
    const float v = src[si];
    vals[u] = v;
    ss += v * v;
  }
  red[tid] = ss;
  __syncthreads();
  for (int s = 128; s > 0; s >>= 1) {
    if (tid < s) red[tid] += red[tid + s];
    __syncthreads();
  }
  const float inv = 1.0f / fmaxf(sqrtf(red[0]), 1e-12f);
  __bf16* qrow = qn + (long long)row * Dd;
  __bf16* rrow = raw ? raw + (long long)row * Dd : nullptr;
#pragma unroll
  for (int u = 0; u < 8; ++u) {
    const int d = u * 256 + tid;
    qrow[d] = (__bf16)(vals[u] * inv);
    if (rrow) rrow[d] = (__bf16)vals[u];
  }
}

// ---------------------------------------------------------------------------
// Row softmax of scaled logits, output bf16 probabilities. Row length = 128.
// ---------------------------------------------------------------------------
__global__ void k_softmax_p(const float* __restrict__ S, __bf16* __restrict__ P,
                            float scale)
{
  __shared__ float red[128];
  const int row = blockIdx.x;
  const int tid = threadIdx.x;
  const float v = S[(long long)row * 128 + tid] * scale;
  red[tid] = v;
  __syncthreads();
  for (int s = 64; s > 0; s >>= 1) {
    if (tid < s) red[tid] = fmaxf(red[tid], red[tid + s]);
    __syncthreads();
  }
  const float m = red[0];
  __syncthreads();
  const float e = __expf(v - m);
  red[tid] = e;
  __syncthreads();
  for (int s = 64; s > 0; s >>= 1) {
    if (tid < s) red[tid] += red[tid + s];
    __syncthreads();
  }
  P[(long long)row * 128 + tid] = (__bf16)(e / red[0]);
}

// ---------------------------------------------------------------------------
// Inverse axial reshape + residual q:  U(b,c,h,w) = O[...] + qsrc(b,c,h,w)
// ---------------------------------------------------------------------------
__global__ void k_unresh_res(const float* __restrict__ O, const float* __restrict__ qsrc,
                             float* __restrict__ U, int axisW)
{
  const long long i = (long long)blockIdx.x * blockDim.x + threadIdx.x;
  if (i >= NEL) return;
  const int w  = (int)(i % Ww);
  const int h  = (int)((i / Ww) % Hh);
  const int ch = (int)((i / HW) % Cc);
  const int b  = (int)(i / CHW);
  const int head = ch >> 4, ci = ch & 15;
  const int row  = axisW ? w : h;
  const int d    = (axisW ? h : w) * CHd + ci;
  const long long oi = (((long long)(b * NHn + head)) * Ld + row) * Dd + d;
  U[i] = O[oi] + qsrc[i];
}

// ---------------------------------------------------------------------------
// Misc elementwise / reduction kernels
// ---------------------------------------------------------------------------
__global__ void k_enh(const float* __restrict__ base, const float* __restrict__ att,
                      const float* __restrict__ rs, float* __restrict__ out)
{
  const long long i = (long long)blockIdx.x * blockDim.x + threadIdx.x;
  if (i < NEL) out[i] = base[i] + rs[0] * att[i];
}

__global__ void k_bn_stats(const float* __restrict__ x, float* __restrict__ mean,
                           float* __restrict__ istd)
{
  __shared__ float r1[256], r2[256];
  const int c = blockIdx.x, tid = threadIdx.x;
  float s = 0.f, ss = 0.f;
  for (int b = 0; b < Bn; ++b) {
    const float* p = x + (long long)(b * Cc + c) * HW;
    for (int i = tid; i < HW; i += 256) { const float v = p[i]; s += v; ss += v * v; }
  }
  r1[tid] = s; r2[tid] = ss;
  __syncthreads();
  for (int s2 = 128; s2 > 0; s2 >>= 1) {
    if (tid < s2) { r1[tid] += r1[tid + s2]; r2[tid] += r2[tid + s2]; }
    __syncthreads();
  }
  if (tid == 0) {
    const float N = (float)(Bn * HW);
    const float mu = r1[0] / N;
    const float var = r2[0] / N - mu * mu;
    mean[c] = mu;
    istd[c] = rsqrtf(var + EPSV);
  }
}

__global__ void k_bn_silu(const float* __restrict__ in, float* __restrict__ out,
                          const float* __restrict__ mean, const float* __restrict__ istd,
                          const float* __restrict__ g, const float* __restrict__ bb)
{
  const long long i = (long long)blockIdx.x * blockDim.x + threadIdx.x;
  if (i >= NEL) return;
  const int c = (int)((i / HW) % Cc);
  out[i] = siluf((in[i] - mean[c]) * istd[c] * g[c] + bb[c]);
}

__global__ void k_dw3(const float* __restrict__ in, const float* __restrict__ w,
                      float* __restrict__ out)
{
  const long long i = (long long)blockIdx.x * blockDim.x + threadIdx.x;
  if (i >= NEL) return;
  const int x = (int)(i % Ww);
  const int y = (int)((i / Ww) % Hh);
  const int c = (int)((i / HW) % Cc);
  const float* base = in + (i - x - (long long)y * Ww);
  const float* wc = w + c * 9;
  float a = 0.f;
  for (int dy = 0; dy < 3; ++dy) {
    const int yy = y + dy - 1;
    if (yy < 0 || yy >= Hh) continue;
    for (int dx = 0; dx < 3; ++dx) {
      const int xx = x + dx - 1;
      if (xx < 0 || xx >= Ww) continue;
      a += wc[dy * 3 + dx] * base[(long long)yy * Ww + xx];
    }
  }
  out[i] = a;
}

__global__ void k_gate_blend(const float* __restrict__ glog, const float* __restrict__ b2,
                             const float* __restrict__ xe, const float* __restrict__ ye,
                             float* __restrict__ fused)
{
  const long long i = (long long)blockIdx.x * blockDim.x + threadIdx.x;
  if (i >= NEL) return;
  const int c = (int)((i / HW) % Cc);
  const float g = sigmoidf(glog[i] + b2[c]);
  fused[i] = g * xe[i] + (1.f - g) * ye[i];
}

__global__ void k_cmean(const float* __restrict__ x, float* __restrict__ out,
                        int ostride, int obase)
{
  __shared__ float red[256];
  const int blk = blockIdx.x;
  const int b = blk / Cc, c = blk % Cc, tid = threadIdx.x;
  const float* p = x + (long long)(b * Cc + c) * HW;
  float s = 0.f;
  for (int i = tid; i < HW; i += 256) s += p[i];
  red[tid] = s;
  __syncthreads();
  for (int s2 = 128; s2 > 0; s2 >>= 1) {
    if (tid < s2) red[tid] += red[tid + s2];
    __syncthreads();
  }
  if (tid == 0) out[(long long)b * ostride + obase + c] = red[0] / (float)HW;
}

// SE channel gate + fusion-strength alpha (tiny MLPs, scalar kernel, 1 block)
__global__ void k_gates(const float* __restrict__ s, const float* __restrict__ pa,
    const float* cgW1, const float* cgB1, const float* cgW2, const float* cgB2,
    const float* alW1, const float* alB1, const float* alW2, const float* alB2,
    float* __restrict__ cg, float* __restrict__ av)
{
  __shared__ float h8[8], a32[32];
  const int tid = threadIdx.x;
  for (int b = 0; b < Bn; ++b) {
    if (tid < 8) {
      float acc = cgB1[tid];
      for (int j = 0; j < Cc; ++j) acc += cgW1[tid * Cc + j] * s[b * Cc + j];
      h8[tid] = siluf(acc);
    }
    if (tid < 32) {
      float acc = alB1[tid];
      for (int j = 0; j < 2 * Cc; ++j) acc += alW1[tid * 2 * Cc + j] * pa[b * 2 * Cc + j];
      a32[tid] = siluf(acc);
    }
    __syncthreads();
    if (tid < Cc) {
      float acc = cgB2[tid];
      for (int j = 0; j < 8; ++j) acc += cgW2[tid * 8 + j] * h8[j];
      cg[b * Cc + tid] = sigmoidf(acc);
    }
    if (tid == 0) {
      float acc = alB2[0];
      for (int j = 0; j < 32; ++j) acc += alW2[j] * a32[j];
      av[b] = sigmoidf(acc);
    }
    __syncthreads();
  }
}

__global__ void k_final_blend(float* __restrict__ fused, const float* __restrict__ cg,
                              const float* __restrict__ av, const float* __restrict__ xe,
                              const float* __restrict__ ye)
{
  const long long i = (long long)blockIdx.x * blockDim.x + threadIdx.x;
  if (i >= NEL) return;
  const int c = (int)((i / HW) % Cc);
  const int b = (int)(i / CHW);
  const float base = 0.5f * (xe[i] + ye[i]);
  const float a = av[b];
  fused[i] = a * (fused[i] * cg[b * Cc + c]) + (1.f - a) * base;
}

__global__ void k_out(const float* __restrict__ ob, const float* __restrict__ mean,
                      const float* __restrict__ istd, const float* __restrict__ g,
                      const float* __restrict__ bb, const float* __restrict__ xe,
                      const float* __restrict__ ye, const float* __restrict__ rsf,
                      float* __restrict__ out)
{
  const long long i = (long long)blockIdx.x * blockDim.x + threadIdx.x;
  if (i >= NEL) return;
  const int c = (int)((i / HW) % Cc);
  const float v = (ob[i] - mean[c]) * istd[c] * g[c] + bb[c];
  out[i] = siluf(v) + rsf[0] * 0.5f * (xe[i] + ye[i]);
}

// ---------------------------------------------------------------------------
extern "C" void kernel_launch(void* const* d_in, const int* in_sizes, int n_in,
                              void* d_out, int out_size, void* d_ws, size_t ws_size,
                              hipStream_t stream)
{
  (void)in_sizes; (void)n_in; (void)out_size; (void)ws_size;
  const float* x      = (const float*)d_in[0];
  const float* y      = (const float*)d_in[1];
  const float* ln_x_w = (const float*)d_in[2];
  const float* ln_x_b = (const float*)d_in[3];
  const float* ln_y_w = (const float*)d_in[4];
  const float* ln_y_b = (const float*)d_in[5];
  // x dir-mix weights: 6..8 (1xk w), 9..11 (1xk b), 12..14 (kx1 w), 15..17 (kx1 b)
  // y dir-mix weights: 18..29 (same pattern)
  const float* proj_x_w = (const float*)d_in[30];
  const float* proj_y_w = (const float*)d_in[31];
  const float* proj_h_w = (const float*)d_in[32];
  const float* proj_w_w = (const float*)d_in[33];
  const float* pg_w1    = (const float*)d_in[34];
  const float* pg_bn1_g = (const float*)d_in[35];
  const float* pg_bn1_b = (const float*)d_in[36];
  const float* pg_dw_w  = (const float*)d_in[37];
  const float* pg_bn2_g = (const float*)d_in[38];
  const float* pg_bn2_b = (const float*)d_in[39];
  const float* pg_w2    = (const float*)d_in[40];
  const float* pg_b2    = (const float*)d_in[41];
  const float* cg_w1    = (const float*)d_in[42];
  const float* cg_b1    = (const float*)d_in[43];
  const float* cg_w2    = (const float*)d_in[44];
  const float* cg_b2    = (const float*)d_in[45];
  const float* al_w1    = (const float*)d_in[46];
  const float* al_b1    = (const float*)d_in[47];
  const float* al_w2    = (const float*)d_in[48];
  const float* al_b2    = (const float*)d_in[49];
  const float* rs_attn  = (const float*)d_in[50];
  const float* rs_fuse  = (const float*)d_in[51];
  const float* out_w    = (const float*)d_in[52];
  const float* out_bn_g = (const float*)d_in[53];
  const float* out_bn_b = (const float*)d_in[54];

  char* wsp = (char*)d_ws;
  size_t off = 0;
  auto alloc = [&](size_t bytes) -> void* {
    void* p = wsp + off;
    off += (bytes + 255) & ~(size_t)255;
    return p;
  };
  float* xn    = (float*)alloc(NEL * 4);
  float* yn    = (float*)alloc(NEL * 4);
  float* mix   = (float*)alloc(NEL * 4);
  float* xd    = (float*)alloc(NEL * 4);
  float* yd    = (float*)alloc(NEL * 4);
  float* xatt  = (float*)alloc(NEL * 4);
  float* yatt  = (float*)alloc(NEL * 4);
  float* xe    = (float*)alloc(NEL * 4);
  float* ye    = (float*)alloc(NEL * 4);
  float* t1    = (float*)alloc(NEL * 4);
  float* t2    = (float*)alloc(NEL * 4);
  float* glog  = (float*)alloc(NEL * 4);
  float* fused = (float*)alloc(NEL * 4);
  float* obuf  = (float*)alloc(NEL * 4);
  float* Sbuf  = (float*)alloc((size_t)Bn * NHn * Ld * Ld * 4);
  float* Obuf  = (float*)alloc(NEL * 4);
  float* Ubuf  = (float*)alloc(NEL * 4);
  __bf16* Qn   = (__bf16*)alloc(NEL * 2);
  __bf16* Kn   = (__bf16*)alloc(NEL * 2);
  __bf16* Vb   = (__bf16*)alloc(NEL * 2);
  __bf16* Pb   = (__bf16*)alloc((size_t)Bn * NHn * Ld * Ld * 2);
  __bf16* bbuf = (__bf16*)alloc(NEL * 2);
  __bf16* xebf = (__bf16*)alloc(NEL * 2);
  __bf16* yebf = (__bf16*)alloc(NEL * 2);
  __bf16* wbuf = (__bf16*)alloc((size_t)Cc * Cc * 2);
  float* mbuf  = (float*)alloc(Cc * 4);
  float* sdbuf = (float*)alloc(Cc * 4);
  float* sbufv = (float*)alloc((size_t)Bn * Cc * 4);
  float* pabuf = (float*)alloc((size_t)Bn * 2 * Cc * 4);
  float* cgbuf = (float*)alloc((size_t)Bn * Cc * 4);
  float* abuf  = (float*)alloc(Bn * 4);

  const dim3 gE((unsigned)((NEL + 255) / 256));
  const dim3 gLN((unsigned)(Bn * HW / 256));
  const dim3 gW((Cc * Cc + 255) / 256);
  const dim3 gRows((unsigned)(Bn * NHn * Ld));
  const dim3 gGemmC(HW / 64, 1, Bn);           // conv1x1 GEMMs (per-batch)
  const float scale = 0.25f;                   // (C/NH)^-0.5 = 16^-0.5

  // ---- 1. LayerNorms ----
  k_ln4d<<<gLN, 256, 0, stream>>>(x, ln_x_w, ln_x_b, xn);
  k_ln4d<<<gLN, 256, 0, stream>>>(y, ln_y_w, ln_y_b, yn);

  // ---- 2. Directional mixes + projections (WMMA conv1x1) ----
  auto dirproj = [&](const float* zn, int base, const float* projW, float* dst) {
    k_dirmix<<<gE, 256, 0, stream>>>(zn, mix,
        (const float*)d_in[base + 0], (const float*)d_in[base + 1], (const float*)d_in[base + 2],
        (const float*)d_in[base + 3], (const float*)d_in[base + 4], (const float*)d_in[base + 5],
        (const float*)d_in[base + 6], (const float*)d_in[base + 7], (const float*)d_in[base + 8],
        (const float*)d_in[base + 9], (const float*)d_in[base + 10], (const float*)d_in[base + 11]);
    k_cvt_bf16<<<gE, 256, 0, stream>>>(mix, bbuf, NEL);
    k_wcvt<<<gW, 256, 0, stream>>>(projW, wbuf, Cc, Cc, Cc, 0);
    k_gemm_bf16<false, false><<<gGemmC, 256, 0, stream>>>(
        wbuf, bbuf, dst, Cc, HW, Cc, 0, CHW, CHW);
  };
  dirproj(xn, 6, proj_x_w, xd);
  dirproj(yn, 18, proj_y_w, yd);

  // ---- 3. Axial cross attention (4 branches, all matmuls in WMMA) ----
  auto branch = [&](const float* qsrc, const float* kvsrc, int axisW,
                    const float* projW, float* dst, bool acc) {
    k_resh_l2n<<<gRows, 256, 0, stream>>>(qsrc, Qn, (__bf16*)nullptr, axisW);
    k_resh_l2n<<<gRows, 256, 0, stream>>>(kvsrc, Kn, Vb, axisW);
    // S = Qn * Kn^T  (batched over B*NH)
    k_gemm_bf16<true, false><<<dim3(Ld / 64, 1, Bn * NHn), 256, 0, stream>>>(
        Qn, Kn, Sbuf, Ld, Ld, Dd, (long long)Ld * Dd, (long long)Ld * Dd, (long long)Ld * Ld);
    k_softmax_p<<<dim3(Bn * NHn * Ld), 128, 0, stream>>>(Sbuf, Pb, scale);
    // O = P * V
    k_gemm_bf16<false, false><<<dim3(Dd / 64, 1, Bn * NHn), 256, 0, stream>>>(
        Pb, Vb, Obuf, Ld, Dd, Ld, (long long)Ld * Ld, (long long)Ld * Dd, (long long)Ld * Dd);
    k_unresh_res<<<gE, 256, 0, stream>>>(Obuf, qsrc, Ubuf, axisW);
    k_cvt_bf16<<<gE, 256, 0, stream>>>(Ubuf, bbuf, NEL);
    k_wcvt<<<gW, 256, 0, stream>>>(projW, wbuf, Cc, Cc, Cc, 0);
    if (acc)
      k_gemm_bf16<false, true><<<gGemmC, 256, 0, stream>>>(wbuf, bbuf, dst, Cc, HW, Cc, 0, CHW, CHW);
    else
      k_gemm_bf16<false, false><<<gGemmC, 256, 0, stream>>>(wbuf, bbuf, dst, Cc, HW, Cc, 0, CHW, CHW);
  };
  branch(yd, xd, 0, proj_h_w, xatt, false);   // x_att, H axis
  branch(yd, xd, 1, proj_w_w, xatt, true);    // x_att, W axis (accumulate)
  branch(xd, yd, 0, proj_h_w, yatt, false);   // y_att, H axis
  branch(xd, yd, 1, proj_w_w, yatt, true);    // y_att, W axis (accumulate)

  // ---- 4. Enhanced streams ----
  k_enh<<<gE, 256, 0, stream>>>(xn, xatt, rs_attn, xe);
  k_enh<<<gE, 256, 0, stream>>>(yn, yatt, rs_attn, ye);

  // ---- 5. PixelGate ----
  k_cvt_bf16<<<gE, 256, 0, stream>>>(xe, xebf, NEL);
  k_cvt_bf16<<<gE, 256, 0, stream>>>(ye, yebf, NEL);
  k_wcvt<<<gW, 256, 0, stream>>>(pg_w1, wbuf, Cc, Cc, 2 * Cc, 0);
  k_gemm_bf16<false, false><<<gGemmC, 256, 0, stream>>>(wbuf, xebf, t1, Cc, HW, Cc, 0, CHW, CHW);
  k_wcvt<<<gW, 256, 0, stream>>>(pg_w1, wbuf, Cc, Cc, 2 * Cc, Cc);
  k_gemm_bf16<false, true><<<gGemmC, 256, 0, stream>>>(wbuf, yebf, t1, Cc, HW, Cc, 0, CHW, CHW);
  k_bn_stats<<<dim3(Cc), 256, 0, stream>>>(t1, mbuf, sdbuf);
  k_bn_silu<<<gE, 256, 0, stream>>>(t1, t1, mbuf, sdbuf, pg_bn1_g, pg_bn1_b);
  k_dw3<<<gE, 256, 0, stream>>>(t1, pg_dw_w, t2);
  k_bn_stats<<<dim3(Cc), 256, 0, stream>>>(t2, mbuf, sdbuf);
  k_bn_silu<<<gE, 256, 0, stream>>>(t2, t2, mbuf, sdbuf, pg_bn2_g, pg_bn2_b);
  k_cvt_bf16<<<gE, 256, 0, stream>>>(t2, bbuf, NEL);
  k_wcvt<<<gW, 256, 0, stream>>>(pg_w2, wbuf, Cc, Cc, Cc, 0);
  k_gemm_bf16<false, false><<<gGemmC, 256, 0, stream>>>(wbuf, bbuf, glog, Cc, HW, Cc, 0, CHW, CHW);
  k_gate_blend<<<gE, 256, 0, stream>>>(glog, pg_b2, xe, ye, fused);

  // ---- 6. ChannelGate (SE) + FusionStrength alpha ----
  k_cmean<<<dim3(Bn * Cc), 256, 0, stream>>>(fused, sbufv, Cc, 0);
  k_cmean<<<dim3(Bn * Cc), 256, 0, stream>>>(xe, pabuf, 2 * Cc, 0);
  k_cmean<<<dim3(Bn * Cc), 256, 0, stream>>>(ye, pabuf, 2 * Cc, Cc);
  k_gates<<<dim3(1), 256, 0, stream>>>(sbufv, pabuf, cg_w1, cg_b1, cg_w2, cg_b2,
                                       al_w1, al_b1, al_w2, al_b2, cgbuf, abuf);
  k_final_blend<<<gE, 256, 0, stream>>>(fused, cgbuf, abuf, xe, ye);

  // ---- 7. Output projection + BN + SiLU + residual ----
  k_cvt_bf16<<<gE, 256, 0, stream>>>(fused, bbuf, NEL);
  k_wcvt<<<gW, 256, 0, stream>>>(out_w, wbuf, Cc, Cc, Cc, 0);
  k_gemm_bf16<false, false><<<gGemmC, 256, 0, stream>>>(wbuf, bbuf, obuf, Cc, HW, Cc, 0, CHW, CHW);
  k_bn_stats<<<dim3(Cc), 256, 0, stream>>>(obuf, mbuf, sdbuf);
  k_out<<<gE, 256, 0, stream>>>(obuf, mbuf, sdbuf, out_bn_g, out_bn_b, xe, ye,
                                rs_fuse, (float*)d_out);
}
